// WindowAttention_1846835937295
// MI455X (gfx1250) — matrix operand order repaired and checked by reading, not compiled, not run on hardware
//
#include <hip/hip_runtime.h>

typedef __attribute__((ext_vector_type(16))) _Float16 v16h;
typedef __attribute__((ext_vector_type(8)))  _Float16 v8h;
typedef __attribute__((ext_vector_type(8)))  float    v8f;

#define WDIM   128
#define WN     49
#define NPAD   64
#define WHEADS 4
#define HD     32
#define WSCALE 0.17677669529663687f   /* 32^-0.5 */
#define WNW    64
#define NEG_BIG (-1.0e30f)

static __device__ __forceinline__ v8f zero8() {
  v8f z;
#pragma unroll
  for (int i = 0; i < 8; ++i) z[i] = 0.0f;
  return z;
}

static __device__ __forceinline__ v16h mk16(v8h lo, v8h hi) {
  return __builtin_shufflevector(lo, hi, 0,1,2,3,4,5,6,7,8,9,10,11,12,13,14,15);
}

// A-fragment (16x32 f16): lane row fixed; 16 contiguous halves at kcol+16*hl
// (K-contiguous, matching B).
static __device__ __forceinline__ v16h ldsA(const _Float16* base, int row, int stride,
                                            int kcol, int hl) {
  const v8h* p0 = (const v8h*)(base + row * stride + kcol + 16 * hl);
  const v8h* p1 = (const v8h*)(base + row * stride + kcol + 16 * hl + 8);
  return mk16(*p0, *p1);
}

// B-fragment (32x16 f16): 16 contiguous halves (K-contiguous, matching the A-fragment).
static __device__ __forceinline__ v16h ldB16(const _Float16* p) {
  const v8h* q = (const v8h*)p;
  return mk16(q[0], q[1]);
}

#define WMMA_F16(a, b, c) \
  __builtin_amdgcn_wmma_f32_16x16x32_f16(false, (a), false, (b), (short)0, (c), false, false)

// ---------------------------------------------------------------------------
// Prep: fp32 -> f16 weight conversion + per-head relative-bias table gather.
// ---------------------------------------------------------------------------
__global__ __launch_bounds__(256)
void swin_prep_kernel(const float* __restrict__ qkv_w, const float* __restrict__ proj_w,
                      const float* __restrict__ bias_table, const int* __restrict__ rel_index,
                      _Float16* __restrict__ qwh, _Float16* __restrict__ pwh,
                      float* __restrict__ relb) {
  int t = blockIdx.x * blockDim.x + threadIdx.x;
  if (t < 3 * WDIM * WDIM) qwh[t] = (_Float16)qkv_w[t];
  if (t < WDIM * WDIM)     pwh[t] = (_Float16)proj_w[t];
  if (t < WHEADS * WN * WN) {
    int h = t / (WN * WN);
    int i = t % (WN * WN);
    relb[t] = bias_table[rel_index[i] * WHEADS + h];
  }
}

// ---------------------------------------------------------------------------
// Fused window attention: 1 block = 1 window, 4 wave32 = 4 heads.
// ---------------------------------------------------------------------------
__global__ __launch_bounds__(128)
void swin_attn_kernel(const float* __restrict__ x, const float* __restrict__ mask,
                      const float* __restrict__ qkv_b, const float* __restrict__ proj_b,
                      const _Float16* __restrict__ qwh, const _Float16* __restrict__ pwh,
                      const float* __restrict__ relb, float* __restrict__ out) {
  __shared__ _Float16 xs[NPAD * WDIM];               // 16 KB  x (f16, padded)
  __shared__ _Float16 qks[2 * WHEADS][NPAD * HD];    // 32 KB  q (mat 0) / k (mat 1)
  __shared__ _Float16 vT[WHEADS][HD * NPAD];         // 16 KB  v transposed [d][tok]
  __shared__ _Float16 Ps[WHEADS][NPAD * NPAD];       // 32 KB  softmax probs
  __shared__ _Float16 Os[NPAD * WDIM];               // 16 KB  attention output

  const int b    = blockIdx.x;
  const int tid  = threadIdx.x;
  const int h    = tid >> 5;          // wave id == head id
  const int lane = tid & 31;
  const int lq   = lane & 15;
  const int hl   = lane >> 4;
  const int w    = b & (WNW - 1);     // window-in-image index for the mask

  // ---- stage x (49x128 fp32 -> f16), zero-pad rows 49..63 -----------------
  {
    const float4* xv = (const float4*)(x + (size_t)b * WN * WDIM);
#pragma unroll 4
    for (int i = tid; i < WN * WDIM / 4; i += 128) {
      float4 v = xv[i];
      int o = i * 4;
      xs[o + 0] = (_Float16)v.x; xs[o + 1] = (_Float16)v.y;
      xs[o + 2] = (_Float16)v.z; xs[o + 3] = (_Float16)v.w;
    }
    for (int i = WN * WDIM + tid; i < NPAD * WDIM; i += 128) xs[i] = (_Float16)0.0f;
  }
  __syncthreads();

  // ---- phase 1a: q (mat 0, pre-scaled) and k (mat 1), row-major -----------
  // Weight B-fragments + biases hoisted out of the mi loop (invariant).
#pragma unroll 1
  for (int mat = 0; mat < 2; ++mat) {
    const int obase = mat * WDIM + h * HD;
    const float scl = (mat == 0) ? WSCALE : 1.0f;
    _Float16* dst = qks[mat * WHEADS + h];
    v16h  bw[2][4];
    float bias[2];
#pragma unroll
    for (int ni = 0; ni < 2; ++ni) {
      const int ocol = obase + ni * 16 + lq;
      bias[ni] = qkv_b[ocol];
#pragma unroll
      for (int kk = 0; kk < 4; ++kk)
        bw[ni][kk] = ldB16(qwh + (size_t)ocol * WDIM + kk * 32 + 16 * hl);
    }
#pragma unroll 1
    for (int mi = 0; mi < 4; ++mi) {
      v16h a[4];
#pragma unroll
      for (int kk = 0; kk < 4; ++kk)
        a[kk] = ldsA(xs, mi * 16 + lq, WDIM, kk * 32, hl);
#pragma unroll
      for (int ni = 0; ni < 2; ++ni) {
        v8f acc = zero8();
#pragma unroll
        for (int kk = 0; kk < 4; ++kk)
          acc = WMMA_F16(a[kk], bw[ni][kk], acc);
#pragma unroll
        for (int r = 0; r < 8; ++r) {
          int m = mi * 16 + hl * 8 + r;
          dst[m * HD + ni * 16 + lq] = (_Float16)((acc[r] + bias[ni]) * scl);
        }
      }
    }
  }

  // ---- phase 1b: v (mat 2), stored transposed [d][tok] --------------------
  {
    const int obase = 2 * WDIM + h * HD;
    v16h  bw[2][4];
    float bias[2];
#pragma unroll
    for (int ni = 0; ni < 2; ++ni) {
      const int ocol = obase + ni * 16 + lq;
      bias[ni] = qkv_b[ocol];
#pragma unroll
      for (int kk = 0; kk < 4; ++kk)
        bw[ni][kk] = ldB16(qwh + (size_t)ocol * WDIM + kk * 32 + 16 * hl);
    }
#pragma unroll 1
    for (int mi = 0; mi < 4; ++mi) {
      v16h a[4];
#pragma unroll
      for (int kk = 0; kk < 4; ++kk)
        a[kk] = ldsA(xs, mi * 16 + lq, WDIM, kk * 32, hl);
#pragma unroll
      for (int ni = 0; ni < 2; ++ni) {
        v8f acc = zero8();
#pragma unroll
        for (int kk = 0; kk < 4; ++kk)
          acc = WMMA_F16(a[kk], bw[ni][kk], acc);
        v8h tmp;                                   // contiguous over r -> b128
#pragma unroll
        for (int r = 0; r < 8; ++r) tmp[r] = (_Float16)(acc[r] + bias[ni]);
        *(v8h*)(vT[h] + (ni * 16 + lq) * NPAD + mi * 16 + hl * 8) = tmp;
      }
    }
  }
  // No barrier: this wave only consumes its own q/k/vT (DS in-order per wave).

  // ---- phase 2: S = q k^T + bias + mask ; row softmax in registers --------
  const _Float16* qsh = qks[h];             // q
  const _Float16* ksh = qks[WHEADS + h];    // k
  const float* relbH = relb + h * WN * WN;
  const float* maskW = mask + (size_t)w * WN * WN;
#pragma unroll 1
  for (int mi = 0; mi < 4; ++mi) {
    const v16h aq = ldsA(qsh, mi * 16 + lq, HD, 0, hl);   // shared across ni
    v8f Sv[4];
#pragma unroll
    for (int ni = 0; ni < 4; ++ni) {
      v16h bf = ldB16(ksh + (ni * 16 + lq) * HD + 16 * hl);
      Sv[ni] = WMMA_F16(aq, bf, zero8());
    }
    float rowm[8];
#pragma unroll
    for (int r = 0; r < 8; ++r) rowm[r] = NEG_BIG;
#pragma unroll
    for (int ni = 0; ni < 4; ++ni) {
      const int n = ni * 16 + lq;
#pragma unroll
      for (int r = 0; r < 8; ++r) {
        const int m = mi * 16 + hl * 8 + r;
        float val = Sv[ni][r];
        if (n < WN) { if (m < WN) val += relbH[m * WN + n] + maskW[m * WN + n]; }
        else        { val = NEG_BIG; }                    // kill padded keys
        Sv[ni][r] = val;
        rowm[r] = fmaxf(rowm[r], val);
      }
    }
    // rows live across the 16 lanes of each half-wave -> xor-reduce 1,2,4,8
#pragma unroll
    for (int r = 0; r < 8; ++r) {
      float v = rowm[r];
      v = fmaxf(v, __shfl_xor(v, 1, 32));
      v = fmaxf(v, __shfl_xor(v, 2, 32));
      v = fmaxf(v, __shfl_xor(v, 4, 32));
      v = fmaxf(v, __shfl_xor(v, 8, 32));
      rowm[r] = v;
    }
    float rsum[8];
#pragma unroll
    for (int r = 0; r < 8; ++r) rsum[r] = 0.0f;
#pragma unroll
    for (int ni = 0; ni < 4; ++ni)
#pragma unroll
      for (int r = 0; r < 8; ++r) {
        float e = __expf(Sv[ni][r] - rowm[r]);
        Sv[ni][r] = e;
        rsum[r] += e;
      }
#pragma unroll
    for (int r = 0; r < 8; ++r) {
      float v = rsum[r];
      v += __shfl_xor(v, 1, 32);
      v += __shfl_xor(v, 2, 32);
      v += __shfl_xor(v, 4, 32);
      v += __shfl_xor(v, 8, 32);
      rsum[r] = __frcp_rn(v);
    }
#pragma unroll
    for (int ni = 0; ni < 4; ++ni) {
      const int n = ni * 16 + lq;
#pragma unroll
      for (int r = 0; r < 8; ++r) {
        const int m = mi * 16 + hl * 8 + r;
        Ps[h][m * NPAD + n] = (_Float16)(Sv[ni][r] * rsum[r]);
      }
    }
  }

  // ---- phase 3: O = P v (K = 64 tokens -> 2 WMMA steps) -------------------
#pragma unroll 1
  for (int mi = 0; mi < 4; ++mi) {
    v16h p[2];
#pragma unroll
    for (int kk = 0; kk < 2; ++kk)
      p[kk] = ldsA(Ps[h], mi * 16 + lq, NPAD, kk * 32, hl);
#pragma unroll
    for (int ni = 0; ni < 2; ++ni) {
      v8f acc = zero8();
#pragma unroll
      for (int kk = 0; kk < 2; ++kk) {
        v16h bf = ldB16(vT[h] + (ni * 16 + lq) * NPAD + kk * 32 + 16 * hl);
        acc = WMMA_F16(p[kk], bf, acc);
      }
#pragma unroll
      for (int r = 0; r < 8; ++r) {
        int m = mi * 16 + hl * 8 + r;
        Os[m * WDIM + h * HD + ni * 16 + lq] = (_Float16)acc[r];
      }
    }
  }
  __syncthreads();   // proj reads all heads' Os slices

  // ---- phase 4: out = O @ Wp^T + b ; wave h owns output cols [32h, 32h+32)
  {
    const int ob = h * HD;
    v16h  bw[2][4];
    float pb[2];
#pragma unroll
    for (int ni = 0; ni < 2; ++ni) {
      const int ocol = ob + ni * 16 + lq;
      pb[ni] = proj_b[ocol];
#pragma unroll
      for (int kk = 0; kk < 4; ++kk)
        bw[ni][kk] = ldB16(pwh + (size_t)ocol * WDIM + kk * 32 + 16 * hl);
    }
#pragma unroll 1
    for (int mi = 0; mi < 4; ++mi) {
      v16h a[4];
#pragma unroll
      for (int kk = 0; kk < 4; ++kk)
        a[kk] = ldsA(Os, mi * 16 + lq, WDIM, kk * 32, hl);
#pragma unroll
      for (int ni = 0; ni < 2; ++ni) {
        v8f acc = zero8();
        const int ocol = ob + ni * 16 + lq;
#pragma unroll
        for (int kk = 0; kk < 4; ++kk)
          acc = WMMA_F16(a[kk], bw[ni][kk], acc);
#pragma unroll
        for (int r = 0; r < 8; ++r) {
          const int m = mi * 16 + hl * 8 + r;
          if (m < WN)
            out[(size_t)b * WN * WDIM + m * WDIM + ocol] = acc[r] + pb[ni];
        }
      }
    }
  }
}

extern "C" void kernel_launch(void* const* d_in, const int* in_sizes, int n_in,
                              void* d_out, int out_size, void* d_ws, size_t ws_size,
                              hipStream_t stream) {
  (void)in_sizes; (void)n_in; (void)out_size; (void)ws_size;
  const float* x          = (const float*)d_in[0];
  const float* mask       = (const float*)d_in[1];
  const float* qkv_w      = (const float*)d_in[2];
  const float* qkv_b      = (const float*)d_in[3];
  const float* proj_w     = (const float*)d_in[4];
  const float* proj_b     = (const float*)d_in[5];
  const float* bias_table = (const float*)d_in[6];
  const int*   rel_index  = (const int*)d_in[7];
  float* out = (float*)d_out;

  _Float16* qwh = (_Float16*)d_ws;                       // 3*128*128 halfs
  _Float16* pwh = qwh + 3 * WDIM * WDIM;                 // 128*128 halfs
  float*    relb = (float*)((char*)d_ws + (size_t)(3 * WDIM * WDIM + WDIM * WDIM) * sizeof(_Float16));

  swin_prep_kernel<<<256, 256, 0, stream>>>(qkv_w, proj_w, bias_table, rel_index,
                                            qwh, pwh, relb);
  swin_attn_kernel<<<4096, 128, 0, stream>>>(x, mask, qkv_b, proj_b, qwh, pwh, relb, out);
}
